// SignLLM_39573828666281
// MI455X (gfx1250) — matrix operand order repaired and checked by reading, not compiled, not run on hardware
//
#include <hip/hip_runtime.h>
#include <hip/hip_bf16.h>
#include <math.h>
#include <stdint.h>

#define B_    32
#define T_    512
#define DIN   1024
#define DHID  512
#define DCODE 1024
#define KCB   256
#define ROWS  (B_ * T_)          // 16384

#define USE_ASYNC_COPY 1

typedef unsigned short u16;
typedef __attribute__((ext_vector_type(16))) __bf16   v16bf;
typedef __attribute__((ext_vector_type(8)))  float    v8f;
typedef __attribute__((ext_vector_type(4)))  unsigned int u32x4;

union Frag { v16bf v; u32x4 q[2]; };

__device__ __forceinline__ u16 f2bf(float f) {
    union { float f; unsigned int u; } c; c.f = f;
    unsigned int u = c.u;
    u += 0x7FFFu + ((u >> 16) & 1u);          // round-to-nearest-even
    return (u16)(u >> 16);
}

__device__ __forceinline__ void async_cp16(unsigned lds_off, const u16* gptr) {
#if USE_ASYNC_COPY
    unsigned long long ga = (unsigned long long)(uintptr_t)gptr;
    asm volatile("global_load_async_to_lds_b128 %0, %1, off"
                 :: "v"(lds_off), "v"(ga) : "memory");
#endif
}

// ---------------- epilogue flags ----------------
#define FL_BIAS 1
#define FL_RELU 2
#define FL_F32  4
#define FL_BF16 8
#define FL_HEAD 16

// C[M,N] = A[M,K] @ Bt[N,K]^T, all bf16 inputs, f32 accumulate.
// M mult of 64 (zero-padded rows ok), N mult of 128, K mult of 32.
// Double-buffered LDS; staging via GLOBAL_LOAD_ASYNC_TO_LDS_B128.
template<int EPI>
__global__ __launch_bounds__(256)
void gemm_wmma(const u16* __restrict__ A, const u16* __restrict__ Bt,
               const float* __restrict__ bias,
               float* __restrict__ C, u16* __restrict__ Cb,
               const float* __restrict__ X, float* __restrict__ partial,
               int M, int N, int K)
{
    __shared__ __align__(16) u16 ldsA[2][64][40];    // [buf][m][k], 80B rows
    __shared__ __align__(16) u16 ldsB[2][128][40];   // [buf][n][k]
    __shared__ float red[256];

    const int tid   = threadIdx.x;
    const int lane  = tid & 31;
    const int wave  = tid >> 5;
    const int half  = lane >> 4;
    const int l16   = lane & 15;
    const int mBase = blockIdx.y * 64;
    const int nBase = blockIdx.x * 128;
    const int mw    = (wave & 1) * 32;
    const int nw    = (wave >> 1) * 32;

    // staging assignment: tid -> 1 A chunk + 2 B chunks (16B each) per K-step
    const int ar = tid >> 2, ac = (tid & 3) * 8;            // A: 64 rows x 4 chunks
    const int br0 = tid >> 2,        bc0 = (tid & 3) * 8;   // B ids 0..255
    const int br1 = (tid + 256) >> 2, bc1 = (tid & 3) * 8;  // B ids 256..511

    v8f acc[2][2];
    const v8f vzero = {0.f,0.f,0.f,0.f,0.f,0.f,0.f,0.f};
    acc[0][0] = vzero; acc[0][1] = vzero; acc[1][0] = vzero; acc[1][1] = vzero;

    const int nsteps = K >> 5;

    auto stage = [&](int k0, int buf) {
#if USE_ASYNC_COPY
        unsigned la = (unsigned)(uintptr_t)&ldsA[buf][0][0];
        unsigned lb = (unsigned)(uintptr_t)&ldsB[buf][0][0];
        async_cp16(la + (unsigned)(ar  * 80 + ac  * 2), A  + (size_t)(mBase + ar ) * K + k0 + ac );
        async_cp16(lb + (unsigned)(br0 * 80 + bc0 * 2), Bt + (size_t)(nBase + br0) * K + k0 + bc0);
        async_cp16(lb + (unsigned)(br1 * 80 + bc1 * 2), Bt + (size_t)(nBase + br1) * K + k0 + bc1);
#else
        *(u32x4*)&ldsA[buf][ar ][ac ] = *(const u32x4*)(A  + (size_t)(mBase + ar ) * K + k0 + ac );
        *(u32x4*)&ldsB[buf][br0][bc0] = *(const u32x4*)(Bt + (size_t)(nBase + br0) * K + k0 + bc0);
        *(u32x4*)&ldsB[buf][br1][bc1] = *(const u32x4*)(Bt + (size_t)(nBase + br1) * K + k0 + bc1);
#endif
    };

    stage(0, 0);
    for (int s = 0; s < nsteps; ++s) {
#if USE_ASYNC_COPY
        asm volatile("s_wait_asynccnt 0" ::: "memory");
#endif
        __syncthreads();
        const int cur = s & 1;
        if (s + 1 < nsteps) stage((s + 1) << 5, cur ^ 1);

        Frag a[2], b[2];
        #pragma unroll
        for (int i = 0; i < 2; ++i) {
            // A 16x32 bf16 frag: lane<16 -> K {0..7,16..23}; lane>=16 -> {8..15,24..31}
            const u16* p = &ldsA[cur][mw + i * 16 + l16][half * 8];
            a[i].q[0] = *(const u32x4*)p;
            a[i].q[1] = *(const u32x4*)(p + 16);
        }
        #pragma unroll
        for (int j = 0; j < 2; ++j) {
            // B 32x16 frag: lane = N col; lane<16 -> K {0..7,16..23}; lane>=16 -> {8..15,24..31}
            const u16* p = &ldsB[cur][nw + j * 16 + l16][half * 8];
            b[j].q[0] = *(const u32x4*)p;
            b[j].q[1] = *(const u32x4*)(p + 16);
        }
        #pragma unroll
        for (int i = 0; i < 2; ++i)
            #pragma unroll
            for (int j = 0; j < 2; ++j)
                acc[i][j] = __builtin_amdgcn_wmma_f32_16x16x32_bf16(
                    false, a[i].v, false, b[j].v, (short)0, acc[i][j], false, false);
    }

    // ---- epilogue ----
    if (EPI & FL_HEAD) {
        float s = 0.f;
        #pragma unroll
        for (int i = 0; i < 2; ++i)
            #pragma unroll
            for (int j = 0; j < 2; ++j)
                #pragma unroll
                for (int r = 0; r < 8; ++r) {
                    int m = mBase + mw + i * 16 + half * 8 + r;
                    int n = nBase + nw + j * 16 + l16;
                    float v = acc[i][j][r] + bias[n];
                    float d = v - X[(size_t)m * N + n];
                    s += d * d;
                }
        __syncthreads();
        red[tid] = s; __syncthreads();
        for (int st = 128; st > 0; st >>= 1) {
            if (tid < st) red[tid] += red[tid + st];
            __syncthreads();
        }
        if (tid == 0) partial[blockIdx.y * gridDim.x + blockIdx.x] = red[0];
    } else {
        #pragma unroll
        for (int i = 0; i < 2; ++i)
            #pragma unroll
            for (int j = 0; j < 2; ++j)
                #pragma unroll
                for (int r = 0; r < 8; ++r) {
                    int m = mBase + mw + i * 16 + half * 8 + r;
                    int n = nBase + nw + j * 16 + l16;
                    float v = acc[i][j][r];
                    if (EPI & FL_BIAS) v += bias[n];
                    if (EPI & FL_RELU) v = fmaxf(v, 0.f);
                    if (EPI & FL_F32)  C [(size_t)m * N + n] = v;
                    if (EPI & FL_BF16) Cb[(size_t)m * N + n] = f2bf(v);
                }
    }
}

// -------- f32 -> bf16 elementwise --------
__global__ __launch_bounds__(256)
void cvt_kernel(const float* __restrict__ in, u16* __restrict__ out, int n)
{
    int i = blockIdx.x * 256 + threadIdx.x;
    if (i < n) out[i] = f2bf(in[i]);
}

// -------- f32 [K][N] -> bf16 [N][K] tiled transpose --------
__global__ __launch_bounds__(256)
void transpose_cvt_kernel(const float* __restrict__ in, u16* __restrict__ out,
                          int K, int N)
{
    __shared__ float tile[32][33];
    const int tx = threadIdx.x & 31, ty = threadIdx.x >> 5;   // ty: 0..7
    const int k0 = blockIdx.x * 32, n0 = blockIdx.y * 32;
    #pragma unroll
    for (int i = 0; i < 32; i += 8)
        tile[ty + i][tx] = in[(size_t)(k0 + ty + i) * N + n0 + tx];
    __syncthreads();
    #pragma unroll
    for (int i = 0; i < 32; i += 8)
        out[(size_t)(n0 + ty + i) * K + k0 + tx] = f2bf(tile[tx][ty + i]);
}

// -------- LayerNorm(512)+ReLU: f32 in -> bf16 out, one block per row --------
__global__ __launch_bounds__(256)
void ln_relu_kernel(const float* __restrict__ h, u16* __restrict__ hb,
                    const float* __restrict__ g, const float* __restrict__ b)
{
    __shared__ float red[256];
    const int tid = threadIdx.x;
    const size_t base = (size_t)blockIdx.x * DHID;
    float v0 = h[base + tid], v1 = h[base + tid + 256];
    red[tid] = v0 + v1; __syncthreads();
    for (int s = 128; s > 0; s >>= 1) { if (tid < s) red[tid] += red[tid + s]; __syncthreads(); }
    float mu = red[0] * (1.f / DHID);
    __syncthreads();
    float d0 = v0 - mu, d1 = v1 - mu;
    red[tid] = d0 * d0 + d1 * d1; __syncthreads();
    for (int s = 128; s > 0; s >>= 1) { if (tid < s) red[tid] += red[tid + s]; __syncthreads(); }
    float rs = rsqrtf(red[0] * (1.f / DHID) + 1e-5f);
    hb[base + tid]       = f2bf(fmaxf(d0 * rs * g[tid]       + b[tid],       0.f));
    hb[base + tid + 256] = f2bf(fmaxf(d1 * rs * g[tid + 256] + b[tid + 256], 0.f));
}

// -------- ||codebook_k||^2 --------
__global__ __launch_bounds__(256)
void cnorm_kernel(const float* __restrict__ cb, float* __restrict__ cn)
{
    __shared__ float red[256];
    const int tid = threadIdx.x;
    const float* row = cb + (size_t)blockIdx.x * DCODE;
    float s = 0.f;
    for (int d = tid; d < DCODE; d += 256) { float v = row[d]; s += v * v; }
    red[tid] = s; __syncthreads();
    for (int st = 128; st > 0; st >>= 1) { if (tid < st) red[tid] += red[tid + st]; __syncthreads(); }
    if (tid == 0) cn[blockIdx.x] = red[0];
}

// -------- argmin_k (||c_k||^2 - 2 f.c_k), first-min tie-break --------
__global__ __launch_bounds__(256)
void argmin_kernel(const float* __restrict__ G, const float* __restrict__ cn,
                   int* __restrict__ idx)
{
    __shared__ float vals[256]; __shared__ int inds[256];
    const int k = threadIdx.x;
    vals[k] = cn[k] - 2.f * G[(size_t)blockIdx.x * KCB + k];
    inds[k] = k;
    __syncthreads();
    for (int s = 128; s > 0; s >>= 1) {
        if (k < s) {
            float v2 = vals[k + s];
            if (v2 < vals[k] || (v2 == vals[k] && inds[k + s] < inds[k])) {
                vals[k] = v2; inds[k] = inds[k + s];
            }
        }
        __syncthreads();
    }
    if (k == 0) idx[blockIdx.x] = inds[0];
}

// -------- gather quantized (bf16 out) + partial sum of (q - f)^2 --------
__global__ __launch_bounds__(256)
void gather_vq_kernel(const float* __restrict__ cb, const int* __restrict__ idx,
                      const float* __restrict__ feat, u16* __restrict__ quant,
                      float* __restrict__ partial)
{
    __shared__ float red[256];
    const int tid = threadIdx.x;
    const size_t gid = (size_t)blockIdx.x * 256 + tid;
    const int r = (int)(gid >> 10), d = (int)(gid & 1023);
    float q = cb[(size_t)idx[r] * DCODE + d];
    quant[gid] = f2bf(q);
    float df = q - feat[gid];
    red[tid] = df * df; __syncthreads();
    for (int s = 128; s > 0; s >>= 1) { if (tid < s) red[tid] += red[tid + s]; __syncthreads(); }
    if (tid == 0) partial[blockIdx.x] = red[0];
}

// -------- deterministic partial-sum reduce --------
__global__ __launch_bounds__(256)
void reduce_scale_kernel(const float* __restrict__ partial, int n, float scale,
                         float* __restrict__ out)
{
    __shared__ float red[256];
    const int tid = threadIdx.x;
    float s = 0.f;
    for (int i = tid; i < n; i += 256) s += partial[i];
    red[tid] = s; __syncthreads();
    for (int st = 128; st > 0; st >>= 1) { if (tid < st) red[tid] += red[tid + st]; __syncthreads(); }
    if (tid == 0) *out = red[0] * scale;
}

// -------- xm[b,d] = mean_t aligned[b,t,d] --------
__global__ __launch_bounds__(256)
void mean_t_kernel(const float* __restrict__ aligned, float* __restrict__ xm)
{
    const int gid = blockIdx.x * 256 + threadIdx.x;   // 32*1024
    const int b = gid >> 10, d = gid & 1023;
    float s = 0.f;
    for (int t = 0; t < T_; ++t) s += aligned[((size_t)(b * T_ + t)) * DCODE + d];
    xm[gid] = s * (1.f / T_);
}

// -------- MMD (32x32 gaussian kernels), one block --------
__global__ __launch_bounds__(256)
void mmd_kernel(const float* __restrict__ xm, const float* __restrict__ prior,
                float* __restrict__ mmd_out)
{
    __shared__ float rxx[256], ryy[256], rxy[256];
    const int tid = threadIdx.x;
    float sxx = 0.f, syy = 0.f, sxy = 0.f;
    for (int p = tid; p < B_ * B_; p += 256) {
        int i = p >> 5, j = p & 31;
        float dxx = 0.f, dyy = 0.f, dxy = 0.f;
        for (int d = 0; d < DCODE; ++d) {
            float a  = xm[i * DCODE + d],    bb = xm[j * DCODE + d];
            float pa = prior[i * DCODE + d], pb = prior[j * DCODE + d];
            float t1 = a - bb;  dxx += t1 * t1;
            float t2 = pa - pb; dyy += t2 * t2;
            float t3 = a - pb;  dxy += t3 * t3;
        }
        sxx += expf(-dxx * (1.f / DCODE));
        syy += expf(-dyy * (1.f / DCODE));
        sxy += expf(-dxy * (1.f / DCODE));
    }
    rxx[tid] = sxx; ryy[tid] = syy; rxy[tid] = sxy; __syncthreads();
    for (int s = 128; s > 0; s >>= 1) {
        if (tid < s) { rxx[tid] += rxx[tid + s]; ryy[tid] += ryy[tid + s]; rxy[tid] += rxy[tid + s]; }
        __syncthreads();
    }
    if (tid == 0)
        *mmd_out = (rxx[0] + ryy[0] - 2.f * rxy[0]) * (1.f / (B_ * B_));
}

// -------- GRU gates; writes h (f32 + bf16) and context bf16 --------
__global__ __launch_bounds__(256)
void gru_gate_kernel(const float* __restrict__ gx, const float* __restrict__ gh,
                     float* __restrict__ h, u16* __restrict__ hb,
                     u16* __restrict__ ctxb, int t)
{
    const int gid = blockIdx.x * 256 + threadIdx.x;   // 32*1024
    const int b = gid >> 10, d = gid & 1023;
    const size_t row = (size_t)b * T_ + t;
    float xr = gx[row * 3072 + d];
    float xz = gx[row * 3072 + 1024 + d];
    float xn = gx[row * 3072 + 2048 + d];
    float hr = gh[(size_t)b * 3072 + d];
    float hz = gh[(size_t)b * 3072 + 1024 + d];
    float hn = gh[(size_t)b * 3072 + 2048 + d];
    float r = 1.f / (1.f + expf(-(xr + hr)));
    float z = 1.f / (1.f + expf(-(xz + hz)));
    float n = tanhf(xn + r * hn);
    float hv = h[gid];
    float hnew = (1.f - z) * n + z * hv;
    h[gid] = hnew;
    hb[gid] = f2bf(hnew);
    ctxb[row * DCODE + d] = hnew == hnew ? f2bf(hnew) : f2bf(hnew); // bf16 context
}

__global__ void zero_kernel(float* __restrict__ p, int n)
{
    int i = blockIdx.x * blockDim.x + threadIdx.x;
    if (i < n) p[i] = 0.f;
}

__global__ void combine_kernel(const float* __restrict__ recon, const float* __restrict__ vq,
                               const float* __restrict__ mmd, float* __restrict__ out)
{
    float r = *recon, v = *vq, m = *mmd;
    out[0] = r; out[1] = v; out[2] = m; out[3] = r + v + 0.5f * m;
}

extern "C" void kernel_launch(void* const* d_in, const int* in_sizes, int n_in,
                              void* d_out, int out_size, void* d_ws, size_t ws_size,
                              hipStream_t stream)
{
    (void)in_sizes; (void)n_in; (void)out_size; (void)ws_size;
    const float* x     = (const float*)d_in[0];
    const float* prior = (const float*)d_in[1];
    const float* encW1 = (const float*)d_in[2];
    const float* encb1 = (const float*)d_in[3];
    const float* lng   = (const float*)d_in[4];
    const float* lnb   = (const float*)d_in[5];
    const float* encW2 = (const float*)d_in[6];
    const float* encb2 = (const float*)d_in[7];
    const float* cb    = (const float*)d_in[8];
    const float* pW1   = (const float*)d_in[9];
    const float* pb1   = (const float*)d_in[10];
    const float* pW2   = (const float*)d_in[11];
    const float* pb2   = (const float*)d_in[12];
    const float* Wih   = (const float*)d_in[13];
    const float* Whh   = (const float*)d_in[14];
    const float* bih   = (const float*)d_in[15];
    const float* bhh   = (const float*)d_in[16];
    const float* headW = (const float*)d_in[17];
    const float* headb = (const float*)d_in[18];
    float* out = (float*)d_out;

    char* ws = (char*)d_ws;
    size_t off = 0;
    auto alloc = [&](size_t bytes) -> void* {
        void* p = ws + off;
        off += (bytes + 255) & ~(size_t)255;
        return p;
    };
    // activations
    u16*   x_bf    = (u16*)  alloc((size_t)ROWS * DIN * 2);
    float* h1      = (float*)alloc((size_t)ROWS * DHID * 4);
    u16*   h1bf    = (u16*)  alloc((size_t)ROWS * DHID * 2);    // reused as p_bf
    float* feat    = (float*)alloc((size_t)ROWS * DCODE * 4);   // reused as aligned
    u16*   featbf  = (u16*)  alloc((size_t)ROWS * DCODE * 2);
    float* G       = (float*)alloc((size_t)ROWS * KCB * 4);
    int*   idx     = (int*)  alloc((size_t)ROWS * 4);
    u16*   quantbf = (u16*)  alloc((size_t)ROWS * DCODE * 2);
    float* gx      = (float*)alloc((size_t)ROWS * 3072 * 4);
    u16*   ctxbf   = (u16*)  alloc((size_t)ROWS * DCODE * 2);
    float* h_f32   = (float*)alloc((size_t)64 * DCODE * 4);     // padded 32->64 rows
    u16*   h_bf    = (u16*)  alloc((size_t)64 * DCODE * 2);
    float* gh_buf  = (float*)alloc((size_t)64 * 3072 * 4);
    float* xm      = (float*)alloc((size_t)B_ * DCODE * 4);
    float* cnorm   = (float*)alloc((size_t)KCB * 4);
    float* partA   = (float*)alloc((size_t)65536 * 4);
    float* partB   = (float*)alloc((size_t)2048 * 4);
    float* scal    = (float*)alloc(64);                         // [0]=vq [1]=mmd [2]=recon
    // bf16 transposed weights [N][K]
    u16* encW1t = (u16*)alloc((size_t)DHID * DIN * 2);
    u16* encW2t = (u16*)alloc((size_t)DCODE * DHID * 2);
    u16* cb_bf  = (u16*)alloc((size_t)KCB * DCODE * 2);         // already [N][K]
    u16* pW1t   = (u16*)alloc((size_t)DHID * DCODE * 2);
    u16* pW2t   = (u16*)alloc((size_t)DCODE * DHID * 2);
    u16* Wiht   = (u16*)alloc((size_t)3072 * DCODE * 2);
    u16* Whht   = (u16*)alloc((size_t)3072 * DCODE * 2);
    u16* headWt = (u16*)alloc((size_t)DIN * DCODE * 2);

    // ---- one-time conversions (per call) ----
    cvt_kernel<<<(ROWS * DIN + 255) / 256, 256, 0, stream>>>(x, x_bf, ROWS * DIN);
    cvt_kernel<<<(KCB * DCODE + 255) / 256, 256, 0, stream>>>(cb, cb_bf, KCB * DCODE);
    transpose_cvt_kernel<<<dim3(DIN / 32, DHID / 32), 256, 0, stream>>>(encW1, encW1t, DIN, DHID);
    transpose_cvt_kernel<<<dim3(DHID / 32, DCODE / 32), 256, 0, stream>>>(encW2, encW2t, DHID, DCODE);
    transpose_cvt_kernel<<<dim3(DCODE / 32, DHID / 32), 256, 0, stream>>>(pW1, pW1t, DCODE, DHID);
    transpose_cvt_kernel<<<dim3(DHID / 32, DCODE / 32), 256, 0, stream>>>(pW2, pW2t, DHID, DCODE);
    transpose_cvt_kernel<<<dim3(DCODE / 32, 3072 / 32), 256, 0, stream>>>(Wih, Wiht, DCODE, 3072);
    transpose_cvt_kernel<<<dim3(DCODE / 32, 3072 / 32), 256, 0, stream>>>(Whh, Whht, DCODE, 3072);
    transpose_cvt_kernel<<<dim3(DCODE / 32, DIN / 32), 256, 0, stream>>>(headW, headWt, DCODE, DIN);
    zero_kernel<<<256, 256, 0, stream>>>(h_f32, 64 * DCODE);
    zero_kernel<<<128, 256, 0, stream>>>((float*)h_bf, 32 * DCODE); // 64*1024 u16 = 32*1024 f32

    // ---- encoder ----
    gemm_wmma<FL_BIAS | FL_F32><<<dim3(DHID / 128, ROWS / 64), 256, 0, stream>>>(
        x_bf, encW1t, encb1, h1, nullptr, nullptr, nullptr, ROWS, DHID, DIN);
    ln_relu_kernel<<<ROWS, 256, 0, stream>>>(h1, h1bf, lng, lnb);
    gemm_wmma<FL_BIAS | FL_F32 | FL_BF16><<<dim3(DCODE / 128, ROWS / 64), 256, 0, stream>>>(
        h1bf, encW2t, encb2, feat, featbf, nullptr, nullptr, ROWS, DCODE, DHID);

    // ---- vector quantizer ----
    cnorm_kernel<<<KCB, 256, 0, stream>>>(cb, cnorm);
    gemm_wmma<FL_F32><<<dim3(KCB / 128, ROWS / 64), 256, 0, stream>>>(
        featbf, cb_bf, nullptr, G, nullptr, nullptr, nullptr, ROWS, KCB, DCODE);
    argmin_kernel<<<ROWS, 256, 0, stream>>>(G, cnorm, idx);
    gather_vq_kernel<<<(ROWS * DCODE) / 256, 256, 0, stream>>>(cb, idx, feat, quantbf, partA);
    reduce_scale_kernel<<<1, 256, 0, stream>>>(partA, 65536,
        1.25f / (float)((size_t)ROWS * DCODE), scal + 0);   // vq_loss = (1+0.25)*mse

    // ---- projector + MMD ----
    gemm_wmma<FL_BIAS | FL_RELU | FL_BF16><<<dim3(DHID / 128, ROWS / 64), 256, 0, stream>>>(
        quantbf, pW1t, pb1, nullptr, h1bf, nullptr, nullptr, ROWS, DHID, DCODE);
    gemm_wmma<FL_BIAS | FL_F32><<<dim3(DCODE / 128, ROWS / 64), 256, 0, stream>>>(
        h1bf, pW2t, pb2, feat, nullptr, nullptr, nullptr, ROWS, DCODE, DHID);
    mean_t_kernel<<<(B_ * DCODE) / 256, 256, 0, stream>>>(feat, xm);
    mmd_kernel<<<1, 256, 0, stream>>>(xm, prior, scal + 1);

    // ---- GRU: big input projection, then 512 sequential steps ----
    gemm_wmma<FL_BIAS | FL_F32><<<dim3(3072 / 128, ROWS / 64), 256, 0, stream>>>(
        quantbf, Wiht, bih, gx, nullptr, nullptr, nullptr, ROWS, 3072, DCODE);
    for (int t = 0; t < T_; ++t) {
        gemm_wmma<FL_BIAS | FL_F32><<<dim3(3072 / 128, 1), 256, 0, stream>>>(
            h_bf, Whht, bhh, gh_buf, nullptr, nullptr, nullptr, 64, 3072, DCODE);
        gru_gate_kernel<<<(B_ * DCODE) / 256, 256, 0, stream>>>(gx, gh_buf, h_f32, h_bf, ctxbf, t);
    }

    // ---- head + fused recon loss ----
    gemm_wmma<FL_HEAD | FL_BIAS><<<dim3(DIN / 128, ROWS / 64), 256, 0, stream>>>(
        ctxbf, headWt, headb, nullptr, nullptr, x, partB, ROWS, DIN, DCODE);
    reduce_scale_kernel<<<1, 256, 0, stream>>>(partB, 2048,
        1.f / (float)((size_t)ROWS * DIN), scal + 2);

    combine_kernel<<<1, 1, 0, stream>>>(scal + 2, scal + 0, scal + 1, out);
}
